// AttentiveFusion_29781303230992
// MI455X (gfx1250) — compile-verified, matrix operand order checked by reading, not run on hardware
//
#include <hip/hip_runtime.h>
#include <math.h>

// ---------------------------------------------------------------------------
// AttentiveFusion for MI455X (gfx1250).
// GEMM chain on v_wmma_f32_16x16x32_bf16 with CDNA5 async-copy staging:
//   global_load_async_to_lds_b128  ->  LDS (double buffered, padded)
//   s_wait_asynccnt 0 + barrier    ->  ds_load_b128 fragments -> WMMA
// Block tile 128x256 (8 waves, each 64x64), f32 accumulate, fused epilogues.
// ---------------------------------------------------------------------------

#define BATCH 16384
#define NBR   3
#define DIM   1024
#define NH    8
#define HDIM  128

#define BM 128
#define BN 256
#define BK 32
#define LDSK 40  // padded row stride (elements): 80B -> bank-conflict spread

typedef __bf16 bf16;
typedef __attribute__((ext_vector_type(16))) __bf16 bf16x16;
typedef __attribute__((ext_vector_type(8)))  __bf16 bf16x8;
typedef __attribute__((ext_vector_type(4)))  __bf16 bf16x4;
typedef __attribute__((ext_vector_type(8)))  float  f32x8;

// ---------------------------------------------------------------------------
// f32 -> bf16 cast
// ---------------------------------------------------------------------------
__global__ __launch_bounds__(256) void cast_f32_to_bf16(
    const float* __restrict__ in, bf16* __restrict__ out, size_t n) {
  size_t i = (size_t)blockIdx.x * 256 + threadIdx.x;
  if (i < n) out[i] = (bf16)in[i];
}

// ---------------------------------------------------------------------------
// Async 16B global->LDS copy (ASYNCcnt-tracked, ISA GLOBAL_LOAD_ASYNC_TO_LDS)
// ---------------------------------------------------------------------------
__device__ __forceinline__ void async_copy16(unsigned int lds_addr,
                                             const void* gptr) {
  asm volatile("global_load_async_to_lds_b128 %0, %1, off"
               :
               : "v"(lds_addr), "v"((unsigned long long)(uintptr_t)gptr)
               : "memory");
}
__device__ __forceinline__ void wait_async0() {
  asm volatile("s_wait_asynccnt 0x0" ::: "memory");
}
__device__ __forceinline__ unsigned int lds_addr_of(const void* p) {
  // generic LDS pointer: low 32 bits are the LDS byte offset (ISA aperture map)
  return (unsigned int)(uintptr_t)p;
}

// ---------------------------------------------------------------------------
// bf16 WMMA fragment from an LDS panel row-major [rows][LDSK], per CDNA5
// 16-bit A/B layout (ISA 7.12.2): lane L -> row (L&15); lanes 0-15 hold
// K = {0..7, 16..23}, lanes 16-31 hold K = {8..15, 24..31}.
// ---------------------------------------------------------------------------
__device__ __forceinline__ bf16x16 load_frag_lds(const bf16* __restrict__ base,
                                                 int r, int half) {
  const bf16* p = base + r * LDSK + half * 8;
  bf16x8 lo = *(const bf16x8*)p;         // K = half*8 ..
  bf16x8 hi = *(const bf16x8*)(p + 16);  // K = 16+half*8 ..
  return __builtin_shufflevector(lo, hi, 0, 1, 2, 3, 4, 5, 6, 7,
                                         8, 9, 10, 11, 12, 13, 14, 15);
}

// ---------------------------------------------------------------------------
// C[M,N] = act(A[M,K] @ W[N,K]^T + bias[N]);  A,W row-major bf16.
// Block = 256 threads = 8 waves (2 x 4) on a 128x256 tile; wave tile 64x64.
// Double-buffered async staging of 128x32 A panel + 256x32 W panel per K-step.
// ---------------------------------------------------------------------------
template <int RELU, int OUTF32>
__global__ __launch_bounds__(256) void gemm_bf16_wmma(
    const bf16* __restrict__ A, const bf16* __restrict__ W,
    const float* __restrict__ bias, void* __restrict__ C,
    int M, int N, int K) {
  __shared__ bf16 sA[2][BM * LDSK];  // 2 x 10 KB
  __shared__ bf16 sB[2][BN * LDSK];  // 2 x 20 KB

  const int t    = threadIdx.x;
  const int lane = t & 31;
  const int wv   = t >> 5;   // wave 0..7
  const int wm   = wv >> 2;  // 0..1  (M)
  const int wn   = wv & 3;   // 0..3  (N)
  const int r    = lane & 15;
  const int half = lane >> 4;

  const int nTilesN = N / BN;
  const int bm = blockIdx.x / nTilesN;
  const int bn = blockIdx.x % nTilesN;

  const bf16* Ablk = A + (size_t)bm * BM * K;
  const bf16* Wblk = W + (size_t)bn * BN * K;

  // Staging: 16B chunks; A = 512 chunks (2/thread), B = 1024 chunks (4/thread)
  auto stage = [&](int buf, int k0) {
#pragma unroll
    for (int s = 0; s < 2; ++s) {
      const int c = t + s * 256;
      const int row = c >> 2, kc = c & 3;
      async_copy16(lds_addr_of(&sA[buf][row * LDSK + kc * 8]),
                   Ablk + (size_t)row * K + k0 + kc * 8);
    }
#pragma unroll
    for (int s = 0; s < 4; ++s) {
      const int c = t + s * 256;
      const int row = c >> 2, kc = c & 3;
      async_copy16(lds_addr_of(&sB[buf][row * LDSK + kc * 8]),
                   Wblk + (size_t)row * K + k0 + kc * 8);
    }
  };

  f32x8 acc[4][4];
#pragma unroll
  for (int i = 0; i < 4; ++i)
#pragma unroll
    for (int j = 0; j < 4; ++j) acc[i][j] = {};

  stage(0, 0);

  int buf = 0;
  for (int k0 = 0; k0 < K; k0 += BK, buf ^= 1) {
    wait_async0();     // our panel-chunk copies have landed in LDS
    __syncthreads();   // everyone's have; also fences re-staging of prev buf
    if (k0 + BK < K) stage(buf ^ 1, k0 + BK);

    const bf16* pa = &sA[buf][wm * 64 * LDSK];
    const bf16* pb = &sB[buf][wn * 64 * LDSK];
    bf16x16 a[4], b[4];
#pragma unroll
    for (int i = 0; i < 4; ++i) a[i] = load_frag_lds(pa + i * 16 * LDSK, r, half);
#pragma unroll
    for (int j = 0; j < 4; ++j) b[j] = load_frag_lds(pb + j * 16 * LDSK, r, half);
#pragma unroll
    for (int i = 0; i < 4; ++i)
#pragma unroll
      for (int j = 0; j < 4; ++j)
        acc[i][j] = __builtin_amdgcn_wmma_f32_16x16x32_bf16(
            false, a[i], false, b[j], (short)0, acc[i][j], false, false);
  }

  // C/D layout: VGPR v, lanes 0-15 -> M=v, lanes 16-31 -> M=v+8; N = lane&15.
  const int c = lane & 15;
#pragma unroll
  for (int j = 0; j < 4; ++j) {
    const int   coln = bn * BN + wn * 64 + j * 16 + c;
    const float bi   = bias ? bias[coln] : 0.0f;
#pragma unroll
    for (int i = 0; i < 4; ++i) {
#pragma unroll
      for (int v = 0; v < 8; ++v) {
        const int rown = bm * BM + wm * 64 + i * 16 + v + 8 * half;
        float val = acc[i][j][v] + bi;
        if (RELU) val = fmaxf(val, 0.0f);
        if (OUTF32)
          ((float*)C)[(size_t)rown * N + coln] = val;
        else
          ((bf16*)C)[(size_t)rown * N + coln] = (bf16)val;
      }
    }
  }
}

// ---------------------------------------------------------------------------
// Tiny attention: one wave per (b, h); seq len NB=3, head dim 128.
// ---------------------------------------------------------------------------
__global__ __launch_bounds__(256) void attn_small(
    const bf16* __restrict__ qkv, bf16* __restrict__ O) {
  const int gt   = blockIdx.x * 256 + threadIdx.x;
  const int wave = gt >> 5;
  const int lane = gt & 31;
  const int b    = wave >> 3;
  const int h    = wave & 7;
  if (b >= BATCH) return;

  const bf16* base = qkv + (size_t)b * NBR * 3 * DIM;
  const int   d0   = h * HDIM + lane * 4;

  float q[NBR][4], k[NBR][4], v[NBR][4];
#pragma unroll
  for (int s = 0; s < NBR; ++s) {
    const bf16* p = base + (size_t)s * 3 * DIM;
    bf16x4 qv = *(const bf16x4*)(p + d0);
    bf16x4 kv = *(const bf16x4*)(p + DIM + d0);
    bf16x4 vv = *(const bf16x4*)(p + 2 * DIM + d0);
#pragma unroll
    for (int j = 0; j < 4; ++j) {
      q[s][j] = (float)qv[j];
      k[s][j] = (float)kv[j];
      v[s][j] = (float)vv[j];
    }
  }

  float sc[NBR][NBR];
#pragma unroll
  for (int qi = 0; qi < NBR; ++qi)
#pragma unroll
    for (int ki = 0; ki < NBR; ++ki) {
      float part = 0.0f;
#pragma unroll
      for (int j = 0; j < 4; ++j) part += q[qi][j] * k[ki][j];
#pragma unroll
      for (int off = 16; off > 0; off >>= 1) part += __shfl_xor(part, off, 32);
      sc[qi][ki] = part;
    }

  const float scale = 0.08838834764831845f;  // 1/sqrt(128)
  bf16* orow = O + (size_t)b * NBR * DIM + h * HDIM + lane * 4;
#pragma unroll
  for (int qi = 0; qi < NBR; ++qi) {
    float s0 = sc[qi][0] * scale, s1 = sc[qi][1] * scale, s2 = sc[qi][2] * scale;
    float m  = fmaxf(s0, fmaxf(s1, s2));
    float e0 = __expf(s0 - m), e1 = __expf(s1 - m), e2 = __expf(s2 - m);
    float inv = 1.0f / (e0 + e1 + e2);
    float w0 = e0 * inv, w1 = e1 * inv, w2 = e2 * inv;
    bf16x4 ov;
#pragma unroll
    for (int j = 0; j < 4; ++j)
      ov[j] = (bf16)(w0 * v[0][j] + w1 * v[1][j] + w2 * v[2][j]);
    *(bf16x4*)(orow + (size_t)qi * DIM) = ov;
  }
}

// ---------------------------------------------------------------------------
// Gate: logits = h2 @ wg3^T + b -> softmax over NB -> weighted branch sum.
// ---------------------------------------------------------------------------
__global__ __launch_bounds__(128) void gate_kernel(
    const bf16* __restrict__ h2, const bf16* __restrict__ wg3,
    const float* __restrict__ wg3_b, const bf16* __restrict__ attended,
    bf16* __restrict__ weighted) {
  const int b = blockIdx.x;
  const int t = threadIdx.x;
  __shared__ float red[3][128];
  __shared__ float wsh[3];

  float p0 = 0.0f, p1 = 0.0f, p2 = 0.0f;
  for (int j = t; j < 512; j += 128) {
    float hv = (float)h2[(size_t)b * 512 + j];
    p0 += hv * (float)wg3[j];
    p1 += hv * (float)wg3[512 + j];
    p2 += hv * (float)wg3[1024 + j];
  }
  red[0][t] = p0; red[1][t] = p1; red[2][t] = p2;
  __syncthreads();
  for (int s = 64; s > 0; s >>= 1) {
    if (t < s) {
      red[0][t] += red[0][t + s];
      red[1][t] += red[1][t + s];
      red[2][t] += red[2][t + s];
    }
    __syncthreads();
  }
  if (t == 0) {
    float l0 = red[0][0] + wg3_b[0];
    float l1 = red[1][0] + wg3_b[1];
    float l2 = red[2][0] + wg3_b[2];
    float m  = fmaxf(l0, fmaxf(l1, l2));
    float e0 = __expf(l0 - m), e1 = __expf(l1 - m), e2 = __expf(l2 - m);
    float inv = 1.0f / (e0 + e1 + e2);
    wsh[0] = e0 * inv; wsh[1] = e1 * inv; wsh[2] = e2 * inv;
  }
  __syncthreads();
  const float w0 = wsh[0], w1 = wsh[1], w2 = wsh[2];
  const bf16* att = attended + (size_t)b * NBR * DIM;
  for (int d = t; d < DIM; d += 128) {
    float val = w0 * (float)att[d] + w1 * (float)att[DIM + d] +
                w2 * (float)att[2 * DIM + d];
    weighted[(size_t)b * DIM + d] = (bf16)val;
  }
}

// ---------------------------------------------------------------------------
// Row LayerNorm (+optional ReLU).  One block (256 threads) per row.
// ---------------------------------------------------------------------------
template <int RELU, int OUT32>
__global__ __launch_bounds__(256) void layernorm_kernel(
    const float* __restrict__ in, const float* __restrict__ g,
    const float* __restrict__ bta, void* __restrict__ out, int C) {
  const int row = blockIdx.x;
  const int t   = threadIdx.x;
  const float* x = in + (size_t)row * C;
  float s = 0.0f, s2 = 0.0f;
  for (int c = t; c < C; c += 256) {
    float v = x[c];
    s += v;
    s2 += v * v;
  }
  __shared__ float rs[256], rs2[256];
  rs[t] = s; rs2[t] = s2;
  __syncthreads();
  for (int k = 128; k > 0; k >>= 1) {
    if (t < k) { rs[t] += rs[t + k]; rs2[t] += rs2[t + k]; }
    __syncthreads();
  }
  const float mean = rs[0] / C;
  const float var  = rs2[0] / C - mean * mean;
  const float inv  = rsqrtf(var + 1e-5f);
  for (int c = t; c < C; c += 256) {
    float v = (x[c] - mean) * inv * g[c] + bta[c];
    if (RELU) v = fmaxf(v, 0.0f);
    if (OUT32)
      ((float*)out)[(size_t)row * C + c] = v;
    else
      ((bf16*)out)[(size_t)row * C + c] = (bf16)v;
  }
}

// ---------------------------------------------------------------------------
// Host-side launch helpers
// ---------------------------------------------------------------------------
static void launch_cast(const float* in, bf16* out, size_t n, hipStream_t s) {
  unsigned blocks = (unsigned)((n + 255) / 256);
  cast_f32_to_bf16<<<blocks, 256, 0, s>>>(in, out, n);
}

static void launch_gemm(int relu, int outf32, const bf16* A, const bf16* W,
                        const float* bias, void* C, int M, int N, int K,
                        hipStream_t s) {
  unsigned blocks = (unsigned)((M / BM) * (N / BN));
  if (relu == 0 && outf32 == 0)
    gemm_bf16_wmma<0, 0><<<blocks, 256, 0, s>>>(A, W, bias, C, M, N, K);
  else if (relu == 1 && outf32 == 0)
    gemm_bf16_wmma<1, 0><<<blocks, 256, 0, s>>>(A, W, bias, C, M, N, K);
  else if (relu == 0 && outf32 == 1)
    gemm_bf16_wmma<0, 1><<<blocks, 256, 0, s>>>(A, W, bias, C, M, N, K);
  else
    gemm_bf16_wmma<1, 1><<<blocks, 256, 0, s>>>(A, W, bias, C, M, N, K);
}

extern "C" void kernel_launch(void* const* d_in, const int* in_sizes, int n_in,
                              void* d_out, int out_size, void* d_ws,
                              size_t ws_size, hipStream_t stream) {
  (void)in_sizes; (void)n_in; (void)out_size; (void)ws_size;

  const float* x     = (const float*)d_in[0];
  const float* in_w  = (const float*)d_in[1];
  const float* in_b  = (const float*)d_in[2];
  const float* out_w = (const float*)d_in[3];
  const float* out_b = (const float*)d_in[4];
  const float* wg1_w = (const float*)d_in[5];
  const float* wg1_b = (const float*)d_in[6];
  const float* wg2_w = (const float*)d_in[7];
  const float* wg2_b = (const float*)d_in[8];
  const float* wg3_w = (const float*)d_in[9];
  const float* wg3_b = (const float*)d_in[10];
  const float* r1_w  = (const float*)d_in[11];
  const float* r1_b  = (const float*)d_in[12];
  const float* ln1_g = (const float*)d_in[13];
  const float* ln1_b = (const float*)d_in[14];
  const float* r2_w  = (const float*)d_in[15];
  const float* r2_b  = (const float*)d_in[16];
  const float* ln2_g = (const float*)d_in[17];
  const float* ln2_b = (const float*)d_in[18];

  const int M3 = BATCH * NBR;  // 49152

  // ---- workspace layout (256-B aligned, aliased regions) -------------------
  size_t off = 0;
  auto take = [&](size_t bytes) {
    size_t o = off;
    off += (bytes + 255) & ~(size_t)255;
    return o;
  };
  char* ws = (char*)d_ws;
  size_t oWqkv = take((size_t)3072 * 1024 * 2);
  size_t oWout = take((size_t)1024 * 1024 * 2);
  size_t oWg1  = take((size_t)1024 * 3072 * 2);
  size_t oWg2  = take((size_t)512 * 1024 * 2);
  size_t oWg3  = take((size_t)3 * 512 * 2);
  size_t oWr1  = take((size_t)2048 * 1024 * 2);
  size_t oWr2  = take((size_t)1024 * 2048 * 2);
  size_t oATT  = take((size_t)M3 * DIM * 2);   // attended bf16
  size_t oO    = take((size_t)M3 * DIM * 2);   // attn output bf16
  size_t oRegA = take((size_t)M3 * DIM * 2);   // Xb; later H1/H2/WGT
  size_t oRegB = take((size_t)M3 * 3072 * 2);  // QKV; later HPre/LN1/OutPre

  bf16* Wqkv = (bf16*)(ws + oWqkv);
  bf16* Wout = (bf16*)(ws + oWout);
  bf16* Wg1  = (bf16*)(ws + oWg1);
  bf16* Wg2  = (bf16*)(ws + oWg2);
  bf16* Wg3  = (bf16*)(ws + oWg3);
  bf16* Wr1  = (bf16*)(ws + oWr1);
  bf16* Wr2  = (bf16*)(ws + oWr2);
  bf16* ATT  = (bf16*)(ws + oATT);
  bf16* Ob   = (bf16*)(ws + oO);
  // Region A: Xb (100.7 MB) dead after QKV GEMM -> reuse for H1/H2/WGT.
  bf16* Xb  = (bf16*)(ws + oRegA);
  bf16* H1  = (bf16*)(ws + oRegA);                     // 33.5 MB
  bf16* H2  = (bf16*)(ws + oRegA + (size_t)33554432);  // 16.8 MB
  bf16* WGT = (bf16*)(ws + oRegA + (size_t)50331648);  // 33.5 MB
  // Region B: QKV (302 MB) dead after attention -> reuse for refiner.
  bf16*  QKV    = (bf16*)(ws + oRegB);
  float* HPre   = (float*)(ws + oRegB);                      // 134.2 MB
  bf16*  LN1    = (bf16*)(ws + oRegB + (size_t)134217728);   // 67.1 MB
  float* OutPre = (float*)(ws + oRegB + (size_t)201326592);  // 67.1 MB

  // ---- stage 0: cast activations + weights to bf16 -------------------------
  launch_cast(x,     Xb,   (size_t)M3 * DIM,    stream);
  launch_cast(in_w,  Wqkv, (size_t)3072 * 1024, stream);
  launch_cast(out_w, Wout, (size_t)1024 * 1024, stream);
  launch_cast(wg1_w, Wg1,  (size_t)1024 * 3072, stream);
  launch_cast(wg2_w, Wg2,  (size_t)512 * 1024,  stream);
  launch_cast(wg3_w, Wg3,  (size_t)3 * 512,     stream);
  launch_cast(r1_w,  Wr1,  (size_t)2048 * 1024, stream);
  launch_cast(r2_w,  Wr2,  (size_t)1024 * 2048, stream);

  // ---- stage 1: QKV projection [49152,1024] x [3072,1024]^T ----------------
  launch_gemm(0, 0, Xb, Wqkv, in_b, QKV, M3, 3072, 1024, stream);

  // ---- stage 2: attention over NB=3 tokens (wave per (b,h)) ----------------
  attn_small<<<(BATCH * NH * 32) / 256, 256, 0, stream>>>(QKV, Ob);

  // ---- stage 3: output projection ------------------------------------------
  launch_gemm(0, 0, Ob, Wout, out_b, ATT, M3, DIM, 1024, stream);

  // ---- stage 4/5: weight-generator MLP (flat view of attended) -------------
  launch_gemm(1, 0, ATT, Wg1, wg1_b, H1, BATCH, 1024, 3072, stream);
  launch_gemm(1, 0, H1,  Wg2, wg2_b, H2, BATCH, 512,  1024, stream);

  // ---- stage 6: gate softmax + weighted branch sum -------------------------
  gate_kernel<<<BATCH, 128, 0, stream>>>(H2, Wg3, wg3_b, ATT, WGT);

  // ---- stage 7: refiner layer 1 + LN + ReLU --------------------------------
  launch_gemm(0, 1, WGT, Wr1, r1_b, HPre, BATCH, 2048, 1024, stream);
  layernorm_kernel<1, 0><<<BATCH, 256, 0, stream>>>(HPre, ln1_g, ln1_b, LN1, 2048);

  // ---- stage 8: refiner layer 2 + final LN ---------------------------------
  launch_gemm(0, 1, LN1, Wr2, r2_b, OutPre, BATCH, 1024, 2048, stream);
  layernorm_kernel<0, 1><<<BATCH, 256, 0, stream>>>(OutPre, ln2_g, ln2_b, d_out, 1024);
}
